// EnhancedTaintFlowGNN_34711925686824
// MI455X (gfx1250) — matrix operand order repaired and checked by reading, not compiled, not run on hardware
//
#include <hip/hip_runtime.h>

typedef __attribute__((ext_vector_type(16))) __bf16 v16bf;
typedef __attribute__((ext_vector_type(8)))  float  v8f;

#define NNODES 150000
#define NPAD   150016          // 32-row-aligned for the GEMM block tile
#define NEDGES 300000
#define NGRAPH 256
#define EMBD   128
#define HID    256
#define NLAB   2

union Frag16 { v16bf v; uint4 q[2]; };

// ---------------- small utility kernels (32-bit shift/mask indexing) -------------

__global__ void k_fill_f32(float* __restrict__ p, float v, unsigned n) {
  unsigned i = blockIdx.x * blockDim.x + threadIdx.x;
  if (i < n) p[i] = v;
}

__global__ void k_deg_edges(const int* __restrict__ ei, float* __restrict__ deg, int E) {
  int e = blockIdx.x * blockDim.x + threadIdx.x;
  if (e < E) atomicAdd(&deg[ei[E + e]], 1.0f);
}

__global__ void k_rsqrt_ip(float* __restrict__ p, unsigned n) {
  unsigned i = blockIdx.x * blockDim.x + threadIdx.x;
  if (i < n) { float d = p[i]; p[i] = d > 0.0f ? rsqrtf(d) : 0.0f; }
}

__global__ void k_embed(const int* __restrict__ ids, const float* __restrict__ emb,
                        float* __restrict__ h, unsigned total) {
  unsigned i = blockIdx.x * blockDim.x + threadIdx.x;
  if (i >= total) return;
  unsigned n = i >> 7, d = i & 127u;              // D = 128
  h[i] = emb[((size_t)ids[n] << 7) + d];
}

// W: [K x Nc] fp32 row-major -> Wt: [Nc x K] bf16 row-major (write coalesced)
template <int KSH>
__global__ void k_wtrans(const float* __restrict__ W, __bf16* __restrict__ Wt, int Nc) {
  constexpr int K = 1 << KSH;
  unsigned i = blockIdx.x * blockDim.x + threadIdx.x;
  if (i >= (unsigned)(K * Nc)) return;
  unsigned n = i >> KSH, k = i & (K - 1);
  Wt[i] = (__bf16)W[(size_t)k * Nc + n];
}

// ---------------- WMMA GEMM ------------------------------------------------------
// C[M x 256] = A[M x K](fp32, lda==K) * Bt[256 x K](bf16)^T
// 256 threads = 8 waves. Block tile: 32 rows x 256 cols.
// Wave w: rows 16*(w>>2), cols 64*(w&3) -> 4 accumulators of 16x16.
// A panel staged to LDS once (fp32->bf16); K-loop is barrier-free.
template <int K>
__global__ __launch_bounds__(256) void k_gemm_bf16(
    const float* __restrict__ A, const __bf16* __restrict__ Bt,
    float* __restrict__ C) {
  __shared__ __bf16 As[32 * K];

  const int t    = threadIdx.x;
  const int row0 = blockIdx.x * 32;

  // stage A panel: 32*K contiguous fp32 -> bf16 LDS (lda == K)
  const float* Abase = A + (size_t)row0 * K;
#pragma unroll
  for (int e = t * 2; e < 32 * K; e += 512) {
    float2 av = *(const float2*)(Abase + e);
    As[e]     = (__bf16)av.x;
    As[e + 1] = (__bf16)av.y;
  }
  __syncthreads();

  const int wave = t >> 5, lane = t & 31;
  const int wr = (wave >> 2) * 16;       // wave row offset: 0 / 16
  const int wc = (wave & 3) * 64;        // wave col offset: 0/64/128/192

  v8f acc0 = {}, acc1 = {}, acc2 = {}, acc3 = {};

  // A fragment addressing (16-bit A 16x32 ISA layout)
  const int r    = lane & 15;
  const int koff = (lane >> 4) * 8;
  const __bf16* As_r = As + (wr + r) * K;
  // B fragment addressing (16-bit B 32x16 layout)
  const int bk = (lane >> 4) * 8;
  const __bf16* B0 = Bt + (size_t)(wc + (lane & 15)) * K;  // cols wc..wc+15
  const __bf16* B1 = B0 + 16 * K;
  const __bf16* B2 = B0 + 32 * K;
  const __bf16* B3 = B0 + 48 * K;

  for (int k0 = 0; k0 < K; k0 += 32) {
    Frag16 af;
    af.q[0] = *(const uint4*)(As_r + k0 + koff);
    af.q[1] = *(const uint4*)(As_r + k0 + 16 + koff);

    Frag16 bf0, bf1, bf2, bf3;
    bf0.q[0] = *(const uint4*)(B0 + k0 + bk);
    bf0.q[1] = *(const uint4*)(B0 + k0 + bk + 16);
    bf1.q[0] = *(const uint4*)(B1 + k0 + bk);
    bf1.q[1] = *(const uint4*)(B1 + k0 + bk + 16);
    bf2.q[0] = *(const uint4*)(B2 + k0 + bk);
    bf2.q[1] = *(const uint4*)(B2 + k0 + bk + 16);
    bf3.q[0] = *(const uint4*)(B3 + k0 + bk);
    bf3.q[1] = *(const uint4*)(B3 + k0 + bk + 16);

    acc0 = __builtin_amdgcn_wmma_f32_16x16x32_bf16(false, af.v, false, bf0.v,
                                                   (short)0, acc0, false, false);
    acc1 = __builtin_amdgcn_wmma_f32_16x16x32_bf16(false, af.v, false, bf1.v,
                                                   (short)0, acc1, false, false);
    acc2 = __builtin_amdgcn_wmma_f32_16x16x32_bf16(false, af.v, false, bf2.v,
                                                   (short)0, acc2, false, false);
    acc3 = __builtin_amdgcn_wmma_f32_16x16x32_bf16(false, af.v, false, bf3.v,
                                                   (short)0, acc3, false, false);
  }

  // D layout: VGPR v -> row = v + 8*(lane>=16), col = lane&15
  const int crow = row0 + wr + (lane >> 4) * 8;
  float* Cp = C + ((size_t)crow << 8) + wc + (lane & 15);
#pragma unroll
  for (int v = 0; v < 8; ++v) {
    Cp[(v << 8) + 0]  = acc0[v];
    Cp[(v << 8) + 16] = acc1[v];
    Cp[(v << 8) + 32] = acc2[v];
    Cp[(v << 8) + 48] = acc3[v];
  }
}

// ---------------- GCN aggregation ------------------------------------------------

// agg[n,j] = m[n,j]*dinv[n]^2 + b[j]   (self-loop + bias fused), float4/thread
__global__ void k_agg_init(const float* __restrict__ m, const float* __restrict__ dinv,
                           const float* __restrict__ b, float* __restrict__ agg,
                           unsigned n4) {
  unsigned i = blockIdx.x * blockDim.x + threadIdx.x;
  if (i >= n4) return;
  unsigned idx = i << 2;
  unsigned n = idx >> 8, j = idx & 255u;          // H = 256
  float di = dinv[n]; float s = di * di;
  float4 mv = *(const float4*)(m + idx);
  float4 bv = *(const float4*)(b + j);
  float4 o;
  o.x = fmaf(mv.x, s, bv.x); o.y = fmaf(mv.y, s, bv.y);
  o.z = fmaf(mv.z, s, bv.z); o.w = fmaf(mv.w, s, bv.w);
  *(float4*)(agg + idx) = o;
}

// agg[dst] += m[src] * (dinv[src]*dinv[dst]); 64 threads per edge row
__global__ void k_scatter(const int* __restrict__ ei, const float* __restrict__ m,
                          const float* __restrict__ dinv, float* __restrict__ agg,
                          unsigned total) {
  unsigned i = blockIdx.x * blockDim.x + threadIdx.x;
  if (i >= total) return;
  unsigned e = i >> 6;                            // H/4 = 64
  unsigned j = (i & 63u) << 2;
  int s = ei[e], d = ei[NEDGES + e];
  float nrm = dinv[s] * dinv[d];
  float4 mv = *(const float4*)(m + (((size_t)s) << 8) + j);
  float* ap = agg + (((size_t)d) << 8) + j;
  atomicAdd(ap + 0, mv.x * nrm);
  atomicAdd(ap + 1, mv.y * nrm);
  atomicAdd(ap + 2, mv.z * nrm);
  atomicAdd(ap + 3, mv.w * nrm);
}

__global__ void k_relu4(float* __restrict__ p, unsigned n4) {
  unsigned i = blockIdx.x * blockDim.x + threadIdx.x;
  if (i >= n4) return;
  float4 v = *(const float4*)(p + ((size_t)i << 2));
  v.x = fmaxf(v.x, 0.f); v.y = fmaxf(v.y, 0.f);
  v.z = fmaxf(v.z, 0.f); v.w = fmaxf(v.w, 0.f);
  *(float4*)(p + ((size_t)i << 2)) = v;
}

// ---------------- pooling + classifier head -------------------------------------

__global__ void k_pool(const float* __restrict__ h, const int* __restrict__ batch,
                       float* __restrict__ gmean, float* __restrict__ gmax,
                       float* __restrict__ counts, unsigned total) {
  unsigned i = blockIdx.x * blockDim.x + threadIdx.x;
  if (i >= total) return;
  unsigned n = i >> 8, j = i & 255u;
  int g = batch[n];
  float v = h[i];
  atomicAdd(&gmean[((unsigned)g << 8) + j], v);
  atomicMax((unsigned int*)&gmax[((unsigned)g << 8) + j], __float_as_uint(v)); // v>=0
  if (j == 0) atomicAdd(&counts[g], 1.0f);
}

__global__ void k_pool_final(const float* __restrict__ gmean, const float* __restrict__ gmax,
                             const float* __restrict__ counts, float* __restrict__ gmat,
                             unsigned total) {
  unsigned i = blockIdx.x * blockDim.x + threadIdx.x;
  if (i >= total) return;
  unsigned g = i >> 9, j = i & 511u;              // 2H = 512
  float c = fmaxf(counts[g], 1.0f);
  gmat[i] = (j < 256u) ? (gmean[(g << 8) + j] / c) : gmax[(g << 8) + (j - 256u)];
}

__global__ __launch_bounds__(256) void k_head(
    const float* __restrict__ hpre, const float* __restrict__ cb0,
    const float* __restrict__ cw1, const float* __restrict__ cb1,
    float* __restrict__ out) {
  __shared__ float sh[HID];
  unsigned g = blockIdx.x, j = threadIdx.x;
  sh[j] = fmaxf(hpre[(g << 8) + j] + cb0[j], 0.0f);
  __syncthreads();
  if (j < NLAB) {
    float s = cb1[j];
    for (int k = 0; k < HID; ++k) s = fmaf(sh[k], cw1[k * NLAB + j], s);
    out[g * NLAB + j] = s;
  }
}

// ---------------- host launcher --------------------------------------------------

extern "C" void kernel_launch(void* const* d_in, const int* in_sizes, int n_in,
                              void* d_out, int out_size, void* d_ws, size_t ws_size,
                              hipStream_t stream) {
  (void)in_sizes; (void)n_in; (void)out_size; (void)ws_size;
  const int*   node_ids = (const int*)d_in[0];
  const int*   ei       = (const int*)d_in[1];   // [2,E]: src=ei[e], dst=ei[E+e]
  const int*   batch    = (const int*)d_in[2];
  const float* emb      = (const float*)d_in[3];
  const float* w0 = (const float*)d_in[4];  const float* b0 = (const float*)d_in[5];
  const float* w1 = (const float*)d_in[6];  const float* b1 = (const float*)d_in[7];
  const float* w2 = (const float*)d_in[8];  const float* b2 = (const float*)d_in[9];
  const float* w3 = (const float*)d_in[10]; const float* b3 = (const float*)d_in[11];
  const float* cw0 = (const float*)d_in[12]; const float* cb0 = (const float*)d_in[13];
  const float* cw1 = (const float*)d_in[14]; const float* cb1 = (const float*)d_in[15];
  float* out = (float*)d_out;

  char* wsb = (char*)d_ws;
  size_t off = 0;
  auto carve = [&](size_t bytes) -> char* {
    size_t o = (off + 255) & ~(size_t)255; off = o + bytes; return wsb + o;
  };
  float* buf_h  = (float*)carve((size_t)NPAD * HID * 4);       // h / agg (aliased)
  float* buf_m  = (float*)carve((size_t)NPAD * HID * 4);       // m = h @ W
  float* dinv   = (float*)carve((size_t)NNODES * 4);           // deg -> rsqrt in place
  float* gmean  = (float*)carve((size_t)NGRAPH * HID * 4);
  float* gmax   = (float*)carve((size_t)NGRAPH * HID * 4);
  float* counts = (float*)carve((size_t)NGRAPH * 4);
  float* gmat   = (float*)carve((size_t)NGRAPH * 2 * HID * 4);
  float* hpre   = (float*)carve((size_t)NGRAPH * HID * 4);
  __bf16* wt0  = (__bf16*)carve((size_t)HID * EMBD * 2);
  __bf16* wt1  = (__bf16*)carve((size_t)HID * HID * 2);
  __bf16* wt2  = (__bf16*)carve((size_t)HID * HID * 2);
  __bf16* wt3  = (__bf16*)carve((size_t)HID * HID * 2);
  __bf16* cwt0 = (__bf16*)carve((size_t)HID * 2 * HID * 2);

  auto blks = [](long n) { return (unsigned)((n + 255) / 256); };

  // degree / normalization
  k_fill_f32<<<blks(NNODES), 256, 0, stream>>>(dinv, 1.0f, NNODES);
  k_deg_edges<<<blks(NEDGES), 256, 0, stream>>>(ei, dinv, NEDGES);
  k_rsqrt_ip<<<blks(NNODES), 256, 0, stream>>>(dinv, NNODES);

  // zero the padded rows (layer-0 128-wide view and 256-wide view)
  k_fill_f32<<<blks((NPAD - NNODES) * EMBD), 256, 0, stream>>>(
      buf_h + (size_t)NNODES * EMBD, 0.0f, (NPAD - NNODES) * EMBD);
  k_fill_f32<<<blks((NPAD - NNODES) * HID), 256, 0, stream>>>(
      buf_h + (size_t)NNODES * HID, 0.0f, (NPAD - NNODES) * HID);

  // embedding gather (buf_h used as [N x 128])
  k_embed<<<blks((long)NNODES * EMBD), 256, 0, stream>>>(node_ids, emb, buf_h,
                                                         (unsigned)(NNODES * EMBD));
  // weights -> bf16, transposed [Hout x K]
  k_wtrans<7><<<blks((long)EMBD * HID), 256, 0, stream>>>(w0, wt0, HID);
  k_wtrans<8><<<blks((long)HID * HID), 256, 0, stream>>>(w1, wt1, HID);
  k_wtrans<8><<<blks((long)HID * HID), 256, 0, stream>>>(w2, wt2, HID);
  k_wtrans<8><<<blks((long)HID * HID), 256, 0, stream>>>(w3, wt3, HID);
  k_wtrans<9><<<blks((long)2 * HID * HID), 256, 0, stream>>>(cw0, cwt0, HID);

  const unsigned n4     = (unsigned)((long)NNODES * HID / 4);
  const unsigned etotal = (unsigned)((long)NEDGES * (HID / 4));
  const dim3 ggrid(NPAD / 32);

  const __bf16* wts[4]   = {wt0, wt1, wt2, wt3};
  const float*  biases[4] = {b0, b1, b2, b3};

  for (int l = 0; l < 4; ++l) {
    if (l == 0)
      k_gemm_bf16<EMBD><<<ggrid, 256, 0, stream>>>(buf_h, wts[0], buf_m);
    else
      k_gemm_bf16<HID><<<ggrid, 256, 0, stream>>>(buf_h, wts[l], buf_m);
    k_agg_init<<<blks(n4), 256, 0, stream>>>(buf_m, dinv, biases[l], buf_h, n4);
    k_scatter<<<blks(etotal), 256, 0, stream>>>(ei, buf_m, dinv, buf_h, etotal);
    k_relu4<<<blks(n4), 256, 0, stream>>>(buf_h, n4);
  }

  // pooling
  k_fill_f32<<<blks((long)NGRAPH * HID), 256, 0, stream>>>(gmean, 0.0f, NGRAPH * HID);
  k_fill_f32<<<blks((long)NGRAPH * HID), 256, 0, stream>>>(gmax, 0.0f, NGRAPH * HID);
  k_fill_f32<<<blks(NGRAPH), 256, 0, stream>>>(counts, 0.0f, NGRAPH);
  k_pool<<<blks((long)NNODES * HID), 256, 0, stream>>>(buf_h, batch, gmean, gmax, counts,
                                                       (unsigned)((long)NNODES * HID));
  k_pool_final<<<blks((long)NGRAPH * 2 * HID), 256, 0, stream>>>(
      gmean, gmax, counts, gmat, (unsigned)(NGRAPH * 2 * HID));

  // classifier: hpre = gmat[256 x 512] @ cw0[512 x 256], then head
  k_gemm_bf16<2 * HID><<<dim3(NGRAPH / 32), 256, 0, stream>>>(gmat, cwt0, hpre);
  k_head<<<NGRAPH, HID, 0, stream>>>(hpre, cb0, cw1, cb1, out);
}
